// CoulombTimeframeAttention_9431748182486
// MI455X (gfx1250) — matrix operand order repaired and checked by reading, not compiled, not run on hardware
//
#include <hip/hip_runtime.h>

// ---------------------------------------------------------------------------
// Types for CDNA5 WMMA
// ---------------------------------------------------------------------------
typedef __attribute__((ext_vector_type(16))) __bf16 v16bf;
typedef __attribute__((ext_vector_type(8)))  float  v8f;

union FragBF {
    v16bf v;
    uint4 q[2];
};

// float -> bf16 (round to nearest even), bf16 -> float
static __device__ __forceinline__ unsigned short f2bf(float f) {
    unsigned int u = __float_as_uint(f);
    u += 0x7FFFu + ((u >> 16) & 1u);
    return (unsigned short)(u >> 16);
}
static __device__ __forceinline__ float bf2f(unsigned short s) {
    return __uint_as_float(((unsigned int)s) << 16);
}

// ---------------------------------------------------------------------------
// Problem constants
// ---------------------------------------------------------------------------
#define NB    8192
#define NTF   11
#define HID   1024
#define DQK   256
#define MROWS (NB * NTF)        // 90112
#define NQKV  1536              // Dqk + Dqk + HID

// GEMM tiling
#define MT  128
#define NT  64
#define KT  64
#define LDA 72   // padded LDS stride (bf16 elems): 144B = 36 banks -> conflict free
#define LDB 72
#define ABUF (MT * LDA)
#define BBUF (NT * LDB)

// workspace layout (bytes, all 256-aligned)
#define OFF_W1    ((size_t)0)                       // [1536][1024] bf16  = 3145728
#define OFF_WO    ((size_t)3145728)                 // [1024][1024] bf16  = 2097152
#define OFF_B1    ((size_t)5242880)                 // [1536] f32         = 6144
#define OFF_BIAS  ((size_t)5249024)                 // [11][11] f32 (pad 512)
#define OFF_HBF   ((size_t)5249536)                 // [90112][1024] bf16 = 184549376
#define OFF_QKV   ((size_t)189798912)               // [90112][1536] bf16 = 276824064
// attended [90112][1024] bf16 aliases OFF_HBF (h_bf16 dead after GEMM-1)
// total ws needed: 466,622,976 bytes (~445 MB)

// ---------------------------------------------------------------------------
// Prep kernels
// ---------------------------------------------------------------------------
__global__ __launch_bounds__(256) void pack_w1_kernel(
    const float* __restrict__ Wq, const float* __restrict__ Wk,
    const float* __restrict__ Wv, const float* __restrict__ bq,
    const float* __restrict__ bk, const float* __restrict__ bv,
    unsigned short* __restrict__ W1, float* __restrict__ b1)
{
    int i = blockIdx.x * 256 + threadIdx.x;      // 1536*1024 total
    int n = i >> 10;
    int k = i & 1023;
    float w;
    if (n < 256)      w = Wq[n * 1024 + k];
    else if (n < 512) w = Wk[(n - 256) * 1024 + k];
    else              w = Wv[(size_t)(n - 512) * 1024 + k];
    W1[i] = f2bf(w);
    if (k == 0) {
        float b;
        if (n < 256)      b = bq[n];
        else if (n < 512) b = bk[n - 256];
        else              b = bv[n - 512];
        b1[n] = b;
    }
}

__global__ __launch_bounds__(256) void pack_wo_kernel(
    const float* __restrict__ Wo, unsigned short* __restrict__ Wo2)
{
    int i = blockIdx.x * 256 + threadIdx.x;      // 1024*1024 total
    Wo2[i] = f2bf(Wo[i]);
}

__global__ __launch_bounds__(256) void conv_h_kernel(
    const float* __restrict__ h, unsigned short* __restrict__ hbf)
{
    size_t i = (size_t)blockIdx.x * 256 + threadIdx.x;   // over float4s
    float4 v = reinterpret_cast<const float4*>(h)[i];
    ushort4 o;
    o.x = f2bf(v.x); o.y = f2bf(v.y); o.z = f2bf(v.z); o.w = f2bf(v.w);
    reinterpret_cast<ushort4*>(hbf)[i] = o;
}

__global__ __launch_bounds__(128) void bias_kernel(
    const float* __restrict__ screen_d, int Kd,
    const float* __restrict__ scale, float* __restrict__ bias11)
{
    int t = threadIdx.x;
    if (t < NTF * NTF) {
        int i = t / NTF, j = t % NTF;
        float r  = fabsf((float)(i - j));
        float w  = 0.f, sg = 1.f;
        for (int k = 0; k < Kd; k++) {
            float d = screen_d[k];
            w  += sg * rsqrtf(d * d + r * r + 1e-6f);
            sg  = -sg;
        }
        bias11[t] = w / (scale[0] + 1e-6f);
    }
}

// ---------------------------------------------------------------------------
// WMMA GEMM:  C[m,n] = sum_k A[m,k] * B[n,k] + bias[n]  (+ resid for MODE 1)
// A: [M][K] bf16 row-major, B: [N][K] bf16 row-major.
// MODE 0: store bf16 (QKV).  MODE 1: store f32 = acc + bias + resid (pre-LN x).
// Double-buffered LDS staging via CDNA5 async global->LDS copies (ASYNCcnt).
// ---------------------------------------------------------------------------
template <int MODE>
__global__ __launch_bounds__(256) void wmma_gemm_kernel(
    const unsigned short* __restrict__ A,
    const unsigned short* __restrict__ B,
    const float* __restrict__ bias,
    const float* __restrict__ resid,       // MODE 1 only, [M][ldc]
    unsigned short* __restrict__ Cb,       // MODE 0
    float* __restrict__ Cf,                // MODE 1
    int K, int ldc, int nBlocksN)
{
    __shared__ unsigned short As[2 * ABUF];
    __shared__ unsigned short Bs[2 * BBUF];

    const int bm = blockIdx.x / nBlocksN;
    const int bn = blockIdx.x % nBlocksN;
    const int m0 = bm * MT;
    const int n0 = bn * NT;

    const int tid  = threadIdx.x;
    const int wave = tid >> 5;
    const int lane = tid & 31;
    const int wm   = wave >> 1;     // 0..3  (M direction)
    const int wn   = wave & 1;      // 0..1  (N direction)
    const int half = lane >> 4;     // 0/1
    const int l16  = lane & 15;

    v8f acc[2][2] = {};

    const int ldrow = tid >> 3;          // 0..31
    const int ldcol = (tid & 7) * 8;     // 8 bf16 = 16B

    // generic LDS pointer low 32 bits == LDS byte offset
    const unsigned as_base = (unsigned)(size_t)(&As[0]);
    const unsigned bs_base = (unsigned)(size_t)(&Bs[0]);

    // Issue async global->LDS copies for one K-tile into buffer `buf`.
    auto issue_tile = [&](int k0, int buf) {
        const unsigned ao =
            as_base + (unsigned)(buf * ABUF + ldrow * LDA + ldcol) * 2u;
        const unsigned bo =
            bs_base + (unsigned)(buf * BBUF + ldrow * LDB + ldcol) * 2u;
#pragma unroll
        for (int i = 0; i < 4; i++) {           // A: 128 x 64 bf16
            const unsigned short* gp =
                A + (size_t)(m0 + ldrow + i * 32) * K + k0 + ldcol;
            const unsigned lo = ao + (unsigned)(i * 32 * LDA) * 2u;
            asm volatile("global_load_async_to_lds_b128 %0, %1, off"
                         :: "v"(lo), "v"(gp) : "memory");
        }
#pragma unroll
        for (int i = 0; i < 2; i++) {           // B: 64 x 64 bf16
            const unsigned short* gp =
                B + (size_t)(n0 + ldrow + i * 32) * K + k0 + ldcol;
            const unsigned lo = bo + (unsigned)(i * 32 * LDB) * 2u;
            asm volatile("global_load_async_to_lds_b128 %0, %1, off"
                         :: "v"(lo), "v"(gp) : "memory");
        }
    };

    issue_tile(0, 0);
    asm volatile("s_wait_asynccnt 0x0" ::: "memory");
    __syncthreads();

    const int nk = K / KT;
    for (int t = 0; t < nk; t++) {
        const int cur = t & 1;
        // prefetch next tile into the other buffer while we compute
        if (t + 1 < nk) issue_tile((t + 1) * KT, cur ^ 1);

        const unsigned short* Ab = &As[cur * ABUF];
        const unsigned short* Bb = &Bs[cur * BBUF];
#pragma unroll
        for (int kk = 0; kk < KT; kk += 32) {
            FragBF a[2], b[2];
            // A frag (16x32 bf16): lanes<16 hold K {0..7,16..23}, lanes>=16 {8..15,24..31}
#pragma unroll
            for (int i = 0; i < 2; i++) {
                const unsigned short* ap =
                    &Ab[(wm * 32 + i * 16 + l16) * LDA + kk + half * 8];
                a[i].q[0] = *reinterpret_cast<const uint4*>(ap);
                a[i].q[1] = *reinterpret_cast<const uint4*>(ap + 16);
            }
            // B frag (32x16 bf16): N = lane%16, lanes<16 hold K {0..7,16..23}, lanes>=16 {8..15,24..31}
#pragma unroll
            for (int j = 0; j < 2; j++) {
                const unsigned short* bp =
                    &Bb[(wn * 32 + j * 16 + l16) * LDB + kk + half * 8];
                b[j].q[0] = *reinterpret_cast<const uint4*>(bp);
                b[j].q[1] = *reinterpret_cast<const uint4*>(bp + 16);
            }
#pragma unroll
            for (int i = 0; i < 2; i++)
#pragma unroll
                for (int j = 0; j < 2; j++)
                    acc[i][j] = __builtin_amdgcn_wmma_f32_16x16x32_bf16(
                        false, a[i].v, false, b[j].v,
                        (short)0, acc[i][j], false, false);
        }
        // next tile's async copies must be LDS-visible before anyone reads them
        asm volatile("s_wait_asynccnt 0x0" ::: "memory");
        __syncthreads();
    }

    // epilogue: C layout -> element e: m = base + half*8 + e, n = n0+tile+l16
#pragma unroll
    for (int j = 0; j < 2; j++) {
        int n  = n0 + wn * 32 + j * 16 + l16;
        float bv = bias[n];
#pragma unroll
        for (int i = 0; i < 2; i++) {
            int mbase = m0 + wm * 32 + i * 16 + half * 8;
#pragma unroll
            for (int e = 0; e < 8; e++) {
                size_t idx = (size_t)(mbase + e) * ldc + n;
                float val = acc[i][j][e] + bv;
                if (MODE == 0) {
                    Cb[idx] = f2bf(val);
                } else {
                    Cf[idx] = val + resid[idx];
                }
            }
        }
    }
}

// ---------------------------------------------------------------------------
// Attention per batch: scores = QK^T/16 + bias; softmax; attended = P @ V
// ---------------------------------------------------------------------------
__global__ __launch_bounds__(128) void attn_kernel(
    const unsigned short* __restrict__ QKV,   // [90112][1536] bf16
    const float* __restrict__ bias11,         // [11][11]
    unsigned short* __restrict__ attended)    // [90112][1024] bf16
{
    __shared__ float          Qs[NTF * DQK];
    __shared__ float          Ks[NTF * DQK];
    __shared__ unsigned short Vs[NTF * HID];
    __shared__ float          Ps[NTF * 16];

    const int b   = blockIdx.x;
    const int tid = threadIdx.x;
    const size_t rb = (size_t)b * NTF * NQKV;

    for (int idx = tid; idx < NTF * DQK; idx += 128) {
        int t = idx >> 8, d = idx & 255;
        Qs[idx] = bf2f(QKV[rb + (size_t)t * NQKV + d]);
        Ks[idx] = bf2f(QKV[rb + (size_t)t * NQKV + DQK + d]);
    }
    for (int idx = tid; idx < NTF * HID; idx += 128) {
        int t = idx >> 10, c = idx & 1023;
        Vs[idx] = QKV[rb + (size_t)t * NQKV + 2 * DQK + c];
    }
    __syncthreads();

    if (tid < NTF * NTF) {
        int q = tid / NTF, k = tid % NTF;
        float s = 0.f;
        for (int d = 0; d < DQK; d++) s += Qs[q * DQK + d] * Ks[k * DQK + d];
        Ps[q * 16 + k] = s * 0.0625f + bias11[q * NTF + k];
    }
    __syncthreads();

    if (tid < NTF) {
        float mx = -1e30f;
        for (int k = 0; k < NTF; k++) mx = fmaxf(mx, Ps[tid * 16 + k]);
        float e[NTF], sum = 0.f;
        for (int k = 0; k < NTF; k++) { e[k] = __expf(Ps[tid * 16 + k] - mx); sum += e[k]; }
        float inv = 1.f / sum;
        for (int k = 0; k < NTF; k++) Ps[tid * 16 + k] = e[k] * inv;
    }
    __syncthreads();

    const size_t ob = (size_t)b * NTF * HID;
#pragma unroll
    for (int j = 0; j < 8; j++) {
        int c = tid + j * 128;
        for (int t = 0; t < NTF; t++) {
            float a = 0.f;
#pragma unroll
            for (int k = 0; k < NTF; k++)
                a += Ps[t * 16 + k] * bf2f(Vs[k * HID + c]);
            attended[ob + (size_t)t * HID + c] = f2bf(a);
        }
    }
}

// ---------------------------------------------------------------------------
// In-place LayerNorm over rows of 1024
// ---------------------------------------------------------------------------
__global__ __launch_bounds__(256) void ln_kernel(
    float* __restrict__ x, const float* __restrict__ gamma,
    const float* __restrict__ beta)
{
    __shared__ float r1[256];
    __shared__ float r2[256];
    const int tid = threadIdx.x;
    float* p = x + (size_t)blockIdx.x * HID;

    float4 v = reinterpret_cast<const float4*>(p)[tid];
    float s  = v.x + v.y + v.z + v.w;
    float sq = v.x * v.x + v.y * v.y + v.z * v.z + v.w * v.w;
    r1[tid] = s; r2[tid] = sq;
    __syncthreads();
    for (int off = 128; off > 0; off >>= 1) {
        if (tid < off) { r1[tid] += r1[tid + off]; r2[tid] += r2[tid + off]; }
        __syncthreads();
    }
    const float inv_n = 1.0f / (float)HID;
    float mu  = r1[0] * inv_n;
    float var = r2[0] * inv_n - mu * mu;
    float rs  = rsqrtf(var + 1e-5f);

    float4 g  = reinterpret_cast<const float4*>(gamma)[tid];
    float4 be = reinterpret_cast<const float4*>(beta)[tid];
    float4 o;
    o.x = (v.x - mu) * rs * g.x + be.x;
    o.y = (v.y - mu) * rs * g.y + be.y;
    o.z = (v.z - mu) * rs * g.z + be.z;
    o.w = (v.w - mu) * rs * g.w + be.w;
    reinterpret_cast<float4*>(p)[tid] = o;
}

// ---------------------------------------------------------------------------
// Launch
// ---------------------------------------------------------------------------
extern "C" void kernel_launch(void* const* d_in, const int* in_sizes, int n_in,
                              void* d_out, int out_size, void* d_ws, size_t ws_size,
                              hipStream_t stream) {
    const float* h        = (const float*)d_in[0];
    const float* Wq       = (const float*)d_in[1];
    const float* bq       = (const float*)d_in[2];
    const float* Wk       = (const float*)d_in[3];
    const float* bk       = (const float*)d_in[4];
    const float* Wv       = (const float*)d_in[5];
    const float* bv       = (const float*)d_in[6];
    const float* Wo       = (const float*)d_in[7];
    const float* bo       = (const float*)d_in[8];
    const float* gamma    = (const float*)d_in[9];
    const float* beta     = (const float*)d_in[10];
    const float* screen_d = (const float*)d_in[11];
    const float* scale    = (const float*)d_in[12];
    float* out            = (float*)d_out;

    char* ws = (char*)d_ws;
    unsigned short* W1     = (unsigned short*)(ws + OFF_W1);
    unsigned short* Wo2    = (unsigned short*)(ws + OFF_WO);
    float*          b1     = (float*)(ws + OFF_B1);
    float*          bias11 = (float*)(ws + OFF_BIAS);
    unsigned short* hbf    = (unsigned short*)(ws + OFF_HBF);
    unsigned short* QKV    = (unsigned short*)(ws + OFF_QKV);
    unsigned short* att    = hbf;   // alias: h_bf16 dead after GEMM-1

    // prep
    pack_w1_kernel<<<(NQKV * HID) / 256, 256, 0, stream>>>(Wq, Wk, Wv, bq, bk, bv, W1, b1);
    pack_wo_kernel<<<(HID * HID) / 256, 256, 0, stream>>>(Wo, Wo2);
    conv_h_kernel<<<(MROWS * HID / 4) / 256, 256, 0, stream>>>(h, hbf);
    bias_kernel<<<1, 128, 0, stream>>>(screen_d, in_sizes[11], scale, bias11);

    // GEMM-1: QKV = h_bf16 @ W1^T + b1  (bf16 out)
    wmma_gemm_kernel<0><<<(MROWS / MT) * (NQKV / NT), 256, 0, stream>>>(
        hbf, W1, b1, nullptr, QKV, nullptr, HID, NQKV, NQKV / NT);

    // attention per batch
    attn_kernel<<<NB, 128, 0, stream>>>(QKV, bias11, att);

    // GEMM-2: x = attended @ Wo^T + bo + h  (f32 out into d_out)
    wmma_gemm_kernel<1><<<(MROWS / MT) * (HID / NT), 256, 0, stream>>>(
        att, Wo2, bo, h, nullptr, out, HID, HID, HID / NT);

    // LayerNorm in place on d_out
    ln_kernel<<<MROWS, 256, 0, stream>>>(out, gamma, beta);
}
